// RNN_76811195121694
// MI455X (gfx1250) — matrix operand order repaired and checked by reading, not compiled, not run on hardware
//
#include <hip/hip_runtime.h>
#include <math.h>

// Problem constants (from reference): T=512, B=64, I=512, H=1024, L=4
constexpr int T = 512, B = 64, I = 512, H = 1024, L = 4;
constexpr int NWG = 32;            // persistent workgroups; each owns H/NWG output cols
constexpr int WG_COLS = H / NWG;   // 32 columns per workgroup (2 x 16-wide WMMA tiles)

typedef __bf16 v16bf __attribute__((ext_vector_type(16)));
typedef __bf16 v8bf  __attribute__((ext_vector_type(8)));
typedef float  v8f   __attribute__((ext_vector_type(8)));

#define DEVINL __device__ __forceinline__

// ---------------------------------------------------------------- f32 -> bf16
__global__ void f32_to_bf16_kernel(const float* __restrict__ s,
                                   __bf16* __restrict__ d, int n) {
  int i = blockIdx.x * blockDim.x + threadIdx.x;
  int stride = gridDim.x * blockDim.x;
  for (; i < n; i += stride) d[i] = (__bf16)s[i];
}

// ------------------------------------------------- wave-level partial GEMM
// Computes (for this wave's 16-column tile at ncol0, K-range [k_begin,k_end))
//   acc[mt] += A[mt*16+row, :] @ W[col, :]^T    via v_wmma_f32_16x16x32_bf16
// A: [64 x lda] bf16 row-major activations, W: [H x ldw] bf16 row-major weights.
// Fragment layouts used here (contiguous k, same order for A and B):
//   A (16x32, 16-bit): lane m=lane&15, half=lane>>4; elem e -> K=half*16+e
//                      (one 32B contiguous load from A row m)
//   B (32x16, 16-bit): lane n=lane&15 is column; elem e -> K=half*16+e
//                      (one 32B contiguous load from W row n)
DEVINL void gemm_acc(const __bf16* __restrict__ A, int lda,
                     const __bf16* __restrict__ W, int ldw,
                     int ncol0, int k_begin, int k_end,
                     int lane, v8f acc[4]) {
  const int cl = lane & 15;
  const int kh = lane >> 4;
  const __bf16* wrow = W + (size_t)(ncol0 + cl) * (size_t)ldw;
  for (int k0 = k_begin; k0 < k_end; k0 += 32) {
    v16bf bf = *(const v16bf*)(wrow + k0 + kh * 16);
#pragma unroll
    for (int mt = 0; mt < 4; ++mt) {
      const __bf16* arow = A + (size_t)(mt * 16 + cl) * (size_t)lda;
      v16bf a = *(const v16bf*)(arow + k0 + kh * 16);
      acc[mt] = __builtin_amdgcn_wmma_f32_16x16x32_bf16(
          false, a, false, bf, (short)0, acc[mt], false, false);
    }
  }
}

// -------------------------------------------------------- device-wide barrier
// Monotonic counter; each WG tracks its own epoch. 4096 barriers * 32 WGs << 2^32.
DEVINL void grid_barrier(unsigned* ctr, unsigned nwg, unsigned& bar) {
  __syncthreads();
  if (threadIdx.x == 0) {
    __threadfence();                       // release prior global writes
    atomicAdd(ctr, 1u);
    bar += 1;
    const unsigned target = bar * nwg;
    while (*(volatile unsigned*)ctr < target)
      __builtin_amdgcn_s_sleep(2);         // polite spin
    __threadfence();                       // acquire others' writes
  }
  __syncthreads();
}

// ------------------------------------------------------------ persistent RNN
// Grid: NWG blocks x 256 threads (8 wave32). Wave w: column-subtile = w&1,
// K-partition = w>>1 (4-way K split, LDS reduction).
__global__ void __launch_bounds__(256, 1)
rnn_persistent(const __bf16* __restrict__ xb,    // [T,B,I]
               const __bf16* __restrict__ Wi0b,  // [H,I]
               const __bf16* __restrict__ WiRb,  // [L-1,H,H]
               const __bf16* __restrict__ Whb,   // [L,H,H]
               const __bf16* __restrict__ Wob,   // [L,H,H]
               const float* __restrict__ bi0,    // [H]
               const float* __restrict__ biR,    // [L-1,H]
               const float* __restrict__ bh,     // [L,H]
               const float* __restrict__ bo,     // [L,H]
               __bf16* __restrict__ hbuf,        // [2,L,B,H] ping-pong hidden
               __bf16* __restrict__ inpb,        // [B,H] inter-layer activation
               __bf16* __restrict__ preb,        // [B,H] pre-activation (bf16)
               float* __restrict__ out,          // [T*B*H] out ++ [L*B*H] hidden
               unsigned* __restrict__ ctr) {
  const int wg   = blockIdx.x;
  const int tid  = threadIdx.x;
  const int lane = tid & 31;
  const int wv   = tid >> 5;
  const int nsub_w = wv & 1;      // which 16-col tile of this WG's 32 cols
  const int kpart  = wv >> 1;     // 4-way K partition
  const int ntile  = wg * WG_COLS + nsub_w * 16;
  unsigned bar = 0;

  __shared__ float red[8 * 4 * 32 * 8];  // [wave][mtile][lane][vgpr] = 32 KB

  for (int t = 0; t < T; ++t) {
    const int p = t & 1;
    const __bf16* hprev = hbuf + (size_t)p * (L * B * H);
    __bf16* hnew        = hbuf + (size_t)(1 - p) * (L * B * H);

    for (int l = 0; l < L; ++l) {
      // ============ phase 1: pre = inp@Wi.T + bi + h@Wh.T + bh ============
      const __bf16* Ain; const __bf16* Wi; const float* bi_l; int Kin;
      if (l == 0) { Ain = xb + (size_t)t * (B * I); Wi = Wi0b; bi_l = bi0; Kin = I; }
      else { Ain = inpb; Wi = WiRb + (size_t)(l - 1) * (H * H);
             bi_l = biR + (l - 1) * H; Kin = H; }

      v8f acc[4];
#pragma unroll
      for (int mt = 0; mt < 4; ++mt)
#pragma unroll
        for (int q = 0; q < 8; ++q) acc[mt][q] = 0.0f;

      const int kc1 = Kin >> 2;
      gemm_acc(Ain, Kin, Wi, Kin, ntile, kpart * kc1, (kpart + 1) * kc1, lane, acc);
      gemm_acc(hprev + (size_t)l * (B * H), H, Whb + (size_t)l * (H * H), H,
               ntile, kpart * (H / 4), (kpart + 1) * (H / 4), lane, acc);

#pragma unroll
      for (int mt = 0; mt < 4; ++mt)
#pragma unroll
        for (int q = 0; q < 8; ++q)
          red[((wv * 4 + mt) * 32 + lane) * 8 + q] = acc[mt][q];
      __syncthreads();

      const float* bhl = bh + l * H;
#pragma unroll
      for (int i = 0; i < 8; ++i) {
        const int o = tid * 8 + i;                 // 2048 outputs per WG
        const int m = o >> 5, j = o & 31;          // row 0..63, local col 0..31
        const int mt = m >> 4, mm = m & 15;
        const int ns = j >> 4, nn = j & 15;
        const int ls = nn + ((mm & 8) ? 16 : 0);   // C/D layout: lane
        const int vg = mm & 7;                     // C/D layout: vgpr
        float s = 0.0f;
#pragma unroll
        for (int kp = 0; kp < 4; ++kp)
          s += red[(((kp * 2 + ns) * 4 + mt) * 32 + ls) * 8 + vg];
        const int col = wg * WG_COLS + j;
        s += bi_l[col] + bhl[col];
        preb[m * H + col] = (__bf16)s;
        const float th = tanhf(s);
        hnew[(size_t)l * (B * H) + m * H + col] = (__bf16)th;
        if (t == T - 1)
          out[(size_t)T * B * H + ((size_t)l * B + m) * H + col] = th;
      }
      grid_barrier(ctr, gridDim.x, bar);

      // ============ phase 2: layer output = tanh(pre@Wo.T + bo) ============
      v8f acc2[4];
#pragma unroll
      for (int mt = 0; mt < 4; ++mt)
#pragma unroll
        for (int q = 0; q < 8; ++q) acc2[mt][q] = 0.0f;

      gemm_acc(preb, H, Wob + (size_t)l * (H * H), H,
               ntile, kpart * (H / 4), (kpart + 1) * (H / 4), lane, acc2);

#pragma unroll
      for (int mt = 0; mt < 4; ++mt)
#pragma unroll
        for (int q = 0; q < 8; ++q)
          red[((wv * 4 + mt) * 32 + lane) * 8 + q] = acc2[mt][q];
      __syncthreads();

      const float* bol = bo + l * H;
#pragma unroll
      for (int i = 0; i < 8; ++i) {
        const int o = tid * 8 + i;
        const int m = o >> 5, j = o & 31;
        const int mt = m >> 4, mm = m & 15;
        const int ns = j >> 4, nn = j & 15;
        const int ls = nn + ((mm & 8) ? 16 : 0);
        const int vg = mm & 7;
        float s = 0.0f;
#pragma unroll
        for (int kp = 0; kp < 4; ++kp)
          s += red[(((kp * 2 + ns) * 4 + mt) * 32 + ls) * 8 + vg];
        const int col = wg * WG_COLS + j;
        s += bol[col];
        const float th = tanhf(s);
        inpb[m * H + col] = (__bf16)th;
        if (l == L - 1)
          out[((size_t)t * B + m) * H + col] = th;
      }
      grid_barrier(ctr, gridDim.x, bar);
    }
  }
}

// ------------------------------------------------------------------- launcher
extern "C" void kernel_launch(void* const* d_in, const int* in_sizes, int n_in,
                              void* d_out, int out_size, void* d_ws, size_t ws_size,
                              hipStream_t stream) {
  (void)in_sizes; (void)n_in; (void)out_size; (void)ws_size;
  const float* x   = (const float*)d_in[0];  // [T,B,I]
  const float* hid = (const float*)d_in[1];  // [L,B,H]
  const float* Wi0 = (const float*)d_in[2];  // [H,I]
  const float* bi0 = (const float*)d_in[3];  // [H]
  const float* WiR = (const float*)d_in[4];  // [L-1,H,H]
  const float* biR = (const float*)d_in[5];  // [L-1,H]
  const float* Wh  = (const float*)d_in[6];  // [L,H,H]
  const float* bh  = (const float*)d_in[7];  // [L,H]
  const float* Wo  = (const float*)d_in[8];  // [L,H,H]
  const float* bo  = (const float*)d_in[9];  // [L,H]
  float* out = (float*)d_out;

  // Workspace carve-out (256B aligned regions), ~57 MB total
  char* w = (char*)d_ws;
  auto carve = [&](size_t bytes) -> char* {
    char* p = w; w += (bytes + 255) & ~(size_t)255; return p;
  };
  unsigned* ctr = (unsigned*)carve(256);
  __bf16* xb    = (__bf16*)carve((size_t)T * B * I * 2);
  __bf16* Wi0b  = (__bf16*)carve((size_t)H * I * 2);
  __bf16* WiRb  = (__bf16*)carve((size_t)(L - 1) * H * H * 2);
  __bf16* Whb   = (__bf16*)carve((size_t)L * H * H * 2);
  __bf16* Wob   = (__bf16*)carve((size_t)L * H * H * 2);
  __bf16* hb    = (__bf16*)carve((size_t)2 * L * B * H * 2);
  __bf16* inpb  = (__bf16*)carve((size_t)B * H * 2);
  __bf16* preb  = (__bf16*)carve((size_t)B * H * 2);

  hipMemsetAsync(ctr, 0, 256, stream);  // barrier counter = 0 every call

  auto cvt = [&](const float* s, __bf16* d, int n) {
    int blocks = (n + 255) / 256; if (blocks > 2048) blocks = 2048;
    hipLaunchKernelGGL(f32_to_bf16_kernel, dim3(blocks), dim3(256), 0, stream,
                       s, d, n);
  };
  cvt(x,   xb,   T * B * I);
  cvt(Wi0, Wi0b, H * I);
  cvt(WiR, WiRb, (L - 1) * H * H);
  cvt(Wh,  Whb,  L * H * H);
  cvt(Wo,  Wob,  L * H * H);
  cvt(hid, hb,   L * B * H);  // parity-0 half; parity-1 is written before read

  hipLaunchKernelGGL(rnn_persistent, dim3(NWG), dim3(256), 0, stream,
                     xb, Wi0b, WiRb, Whb, Wob, bi0, biR, bh, bo,
                     hb, inpb, preb, out, ctr);
}
